// DGL_MPNNLayer_88648124989657
// MI455X (gfx1250) — matrix operand order repaired and checked by reading, not compiled, not run on hardware
//
#include <hip/hip_runtime.h>
#include <hip/hip_bf16.h>

typedef __attribute__((ext_vector_type(16))) __bf16 v16bf;
typedef __attribute__((ext_vector_type(8)))  float  v8f;

#define HID 32
#define EDGE_DIM 16
#define EDGE_HIDDEN 128
#define KSTEPS 129                         // 128 h-steps + 1 folded-b2 step
#define W2SW_ELEMS (KSTEPS * 2 * 32 * 16)  // 132096 bf16 elements
#define W2SW_BYTES (W2SW_ELEMS * 2)        // 264192 bytes (256B multiple)

// ---------------------------------------------------------------------------
// Kernel 1: pre-swizzle W2 (with b2 folded in as 32 extra K-rows) into the
// exact per-lane WMMA B-fragment order, bf16.
// B k order used here (contiguous, A matches; 32x16 bf16, wave32): lanes 0-15: N=lane, K=c (c=0..15);
// lanes 16-31: N=lane-16, K=16+c. Stored as [j][nh][lane][c].
// W2'[kk][o] == W2flat[kk*32 + o] (kk = k*32+i); rows 4096..4127 come from b2.
// ---------------------------------------------------------------------------
__global__ void prep_w2_kernel(const float* __restrict__ W2,
                               const float* __restrict__ b2,
                               __bf16* __restrict__ w2sw) {
    int idx = blockIdx.x * blockDim.x + threadIdx.x;
    if (idx >= W2SW_ELEMS) return;
    int c    = idx & 15;
    int lane = (idx >> 4) & 31;
    int nh   = (idx >> 9) & 1;
    int j    = idx >> 10;                       // 0..128
    int K32  = (lane < 16) ? c : (16 + c);
    int o    = nh * 16 + (lane & 15);
    int kk   = j * 32 + K32;                    // row of W2' [4128 x 32]
    float v  = (kk < EDGE_HIDDEN * HID)
                 ? W2[(size_t)kk * HID + o]
                 : b2[(size_t)(kk - EDGE_HIDDEN * HID) * HID + o];
    w2sw[idx] = (__bf16)v;
}

// ---------------------------------------------------------------------------
// Kernel 2: out[n, o] = bias[o]  (atomics accumulate on top of this)
// ---------------------------------------------------------------------------
__global__ void init_out_kernel(const float* __restrict__ bias,
                                float* __restrict__ out, int n_elems) {
    int idx = blockIdx.x * blockDim.x + threadIdx.x;
    if (idx < n_elems) out[idx] = bias[idx & (HID - 1)];
}

// ---------------------------------------------------------------------------
// Kernel 3: h = relu(ef @ W1 + b1) -> bf16 workspace. One thread per (e,k).
// ---------------------------------------------------------------------------
__global__ void edge_mlp_kernel(const float* __restrict__ ef,
                                const float* __restrict__ W1,
                                const float* __restrict__ b1,
                                __bf16* __restrict__ h, int total) {
    int t = blockIdx.x * blockDim.x + threadIdx.x;
    if (t >= total) return;
    int e = t >> 7;                 // edge
    int k = t & 127;                // hidden unit
    const float* er = ef + (size_t)e * EDGE_DIM;
    float acc = b1[k];
    #pragma unroll
    for (int i = 0; i < EDGE_DIM; ++i)
        acc = fmaf(er[i], W1[i * EDGE_HIDDEN + k], acc);
    h[t] = (__bf16)fmaxf(acc, 0.0f);
}

// ---------------------------------------------------------------------------
// Kernel 4: main fused NNConv GEMM. One wave32 owns 64 edges (4 stacked
// 16x32 A tiles) so each 2KB B fragment pair feeds 8 WMMAs (4x register-level
// B reuse); 8 waves per workgroup march in near-lockstep through the 258KB
// w2sw stream so the active window stays WGP$-resident (cross-wave reuse).
// msg[64,32] = (h (x) x)[64,4128] @ W2'[4128,32] via v_wmma_f32_16x16x32_bf16.
// A fragments are rank-1 outer products built with packed bf16 multiplies.
// ---------------------------------------------------------------------------
__global__ void __launch_bounds__(256)
nnconv_wmma_kernel(const float* __restrict__ nf,
                   const int*   __restrict__ edge_src,
                   const int*   __restrict__ edge_dst,
                   const __bf16* __restrict__ h,
                   const __bf16* __restrict__ w2sw,
                   float* __restrict__ out,
                   int n_super) {
    const int lane = threadIdx.x & 31;
    const int s    = (blockIdx.x * blockDim.x + threadIdx.x) >> 5;
    if (s >= n_super) return;        // uniform per wave -> EXEC all-1s below

    const int row   = lane & 15;
    const int b0i   = (lane < 16) ? 0 : 16;
    const int ebase = s * 64 + row;

    // Gather + convert source features for the 4 tiles (A-layout order):
    // lanes 0-15 hold x indices {0..15}; lanes 16-31 hold {16..31}.
    v16bf xbf[4];
    const __bf16* hr[4];
    #pragma unroll
    for (int tt = 0; tt < 4; ++tt) {
        const int e   = ebase + tt * 16;
        const int src = edge_src[e];
        const float* xr = nf + (size_t)src * HID;
        float4 xa = *(const float4*)(xr + b0i);
        float4 xb = *(const float4*)(xr + b0i + 4);
        float4 xc = *(const float4*)(xr + b0i + 8);
        float4 xd = *(const float4*)(xr + b0i + 12);
        v16bf xv;
        xv[0]  = (__bf16)xa.x; xv[1]  = (__bf16)xa.y;
        xv[2]  = (__bf16)xa.z; xv[3]  = (__bf16)xa.w;
        xv[4]  = (__bf16)xb.x; xv[5]  = (__bf16)xb.y;
        xv[6]  = (__bf16)xb.z; xv[7]  = (__bf16)xb.w;
        xv[8]  = (__bf16)xc.x; xv[9]  = (__bf16)xc.y;
        xv[10] = (__bf16)xc.z; xv[11] = (__bf16)xc.w;
        xv[12] = (__bf16)xd.x; xv[13] = (__bf16)xd.y;
        xv[14] = (__bf16)xd.z; xv[15] = (__bf16)xd.w;
        xbf[tt] = xv;
        hr[tt]  = h + (size_t)e * EDGE_HIDDEN;
    }

    const __bf16* bp = w2sw + lane * 16;   // per-lane B-fragment base

    v8f acc[4][2] = {};

    for (int jc = 0; jc < 8; ++jc) {
        v16bf hch[4];
        #pragma unroll
        for (int tt = 0; tt < 4; ++tt)
            hch[tt] = *(const v16bf*)(hr[tt] + jc * 16);   // 16 h-scalars
        #pragma unroll
        for (int t = 0; t < 16; ++t) {
            const int j = jc * 16 + t;
            v16bf bb0 = *(const v16bf*)(bp + (size_t)(j * 2 + 0) * 512);
            v16bf bb1 = *(const v16bf*)(bp + (size_t)(j * 2 + 1) * 512);
            // Walk one 32KB K-chunk ahead of the consume pointer
            // (global_prefetch_b8; lanes stride 32B so a wave covers ~1KB/step).
            __builtin_prefetch((const void*)(bp + (size_t)((j + 16) * 2 + 0) * 512), 0, 3);
            __builtin_prefetch((const void*)(bp + (size_t)((j + 16) * 2 + 1) * 512), 0, 3);
            #pragma unroll
            for (int tt = 0; tt < 4; ++tt) {
                const __bf16 hb = hch[tt][t];
                v16bf hs;
                #pragma unroll
                for (int c = 0; c < 16; ++c) hs[c] = hb;
                v16bf a = xbf[tt] * hs;          // packed bf16 rank-1 build
                acc[tt][0] = __builtin_amdgcn_wmma_f32_16x16x32_bf16(
                                 false, a, false, bb0, (short)0, acc[tt][0], false, false);
                acc[tt][1] = __builtin_amdgcn_wmma_f32_16x16x32_bf16(
                                 false, a, false, bb1, (short)0, acc[tt][1], false, false);
            }
        }
    }
    // Folded b2 step: j = 128, implicit h = 1  => A fragment is just x.
    {
        v16bf bb0 = *(const v16bf*)(bp + (size_t)(128 * 2 + 0) * 512);
        v16bf bb1 = *(const v16bf*)(bp + (size_t)(128 * 2 + 1) * 512);
        #pragma unroll
        for (int tt = 0; tt < 4; ++tt) {
            acc[tt][0] = __builtin_amdgcn_wmma_f32_16x16x32_bf16(
                             false, xbf[tt], false, bb0, (short)0, acc[tt][0], false, false);
            acc[tt][1] = __builtin_amdgcn_wmma_f32_16x16x32_bf16(
                             false, xbf[tt], false, bb1, (short)0, acc[tt][1], false, false);
        }
    }

    // C/D layout: VGPR g, lanes 0-15: M=g, N=lane; lanes 16-31: M=g+8, N=lane-16.
    const int n  = lane & 15;
    const int mb = (lane < 16) ? 0 : 8;
    #pragma unroll
    for (int tt = 0; tt < 4; ++tt) {
        #pragma unroll
        for (int g = 0; g < 8; ++g) {
            const int eo  = s * 64 + tt * 16 + mb + g;
            const int dst = edge_dst[eo];
            float* op = out + (size_t)dst * HID;
            atomicAdd(op + n,      acc[tt][0][g]);
            atomicAdd(op + 16 + n, acc[tt][1][g]);
        }
    }
}

// ---------------------------------------------------------------------------
extern "C" void kernel_launch(void* const* d_in, const int* in_sizes, int n_in,
                              void* d_out, int out_size, void* d_ws, size_t ws_size,
                              hipStream_t stream) {
    const float* nf   = (const float*)d_in[0];
    const float* ef   = (const float*)d_in[1];
    const int*   esrc = (const int*)  d_in[2];
    const int*   edst = (const int*)  d_in[3];
    const float* W1   = (const float*)d_in[4];
    const float* b1   = (const float*)d_in[5];
    const float* W2   = (const float*)d_in[6];
    const float* b2   = (const float*)d_in[7];
    const float* bias = (const float*)d_in[8];
    float* out = (float*)d_out;

    const int E  = in_sizes[2];           // 200000
    const int NN = in_sizes[0] / HID;     // 50000

    __bf16* w2sw = (__bf16*)d_ws;
    __bf16* hbuf = (__bf16*)((char*)d_ws + W2SW_BYTES);

    // 1) pre-swizzle W2/b2 into WMMA B-fragment order (bf16)
    prep_w2_kernel<<<(W2SW_ELEMS + 255) / 256, 256, 0, stream>>>(W2, b2, w2sw);

    // 2) out = bias (atomics accumulate on top)
    const int n_out = NN * HID;
    init_out_kernel<<<(n_out + 255) / 256, 256, 0, stream>>>(bias, out, n_out);

    // 3) edge MLP layer 1 -> bf16 h
    const int total_h = E * EDGE_HIDDEN;
    edge_mlp_kernel<<<(total_h + 255) / 256, 256, 0, stream>>>(ef, W1, b1, hbuf, total_h);

    // 4) fused rank-1-GEMM + scatter (wave owns 64 edges, 8 waves/block)
    const int n_super = E / 64;                       // 3125
    const int blocks  = (n_super + 7) / 8;            // 391
    nnconv_wmma_kernel<<<blocks, 256, 0, stream>>>(nf, esrc, edst, hbuf, w2sw, out, n_super);
}
